// SparseBlock_60979945669305
// MI455X (gfx1250) — matrix operand order repaired and checked by reading, not compiled, not run on hardware
//
#include <hip/hip_runtime.h>

// ---------------------------------------------------------------------------
// SparseBlock for MI455X (gfx1250): bf16 WMMA pointwise GEMM fused with
// gather depthwise conv + BN (+relu / +skip). Two passes over N, bf16
// intermediates to halve gather bandwidth (the roofline bottleneck).
// ---------------------------------------------------------------------------

#define NPTS 250000
#define CCH  256
#define KNBR 9
#define TM   16          // points per tile
#define NTILES (NPTS / TM)   // 15625

typedef __bf16 v16bf __attribute__((ext_vector_type(16)));
typedef float  v8f   __attribute__((ext_vector_type(8)));

__device__ __forceinline__ unsigned short f2bf(float f) {
  unsigned int u = __float_as_uint(f);
  u += 0x7fffu + ((u >> 16) & 1u);            // round-to-nearest-even
  return (unsigned short)(u >> 16);
}
__device__ __forceinline__ float bf2f(unsigned short h) {
  return __uint_as_float(((unsigned int)h) << 16);
}

// ---- prep: xh = bf16(relu(x)), 8 elems/thread, fully coalesced ------------
__global__ __launch_bounds__(256) void relu_to_bf16(const float* __restrict__ x,
                                                    unsigned short* __restrict__ xh) {
  long long e = ((long long)blockIdx.x * blockDim.x + threadIdx.x) * 8;
  const float4* p = (const float4*)(x + e);
  float4 a = p[0], b = p[1];
  unsigned int w0 = (unsigned int)f2bf(fmaxf(a.x, 0.f)) | ((unsigned int)f2bf(fmaxf(a.y, 0.f)) << 16);
  unsigned int w1 = (unsigned int)f2bf(fmaxf(a.z, 0.f)) | ((unsigned int)f2bf(fmaxf(a.w, 0.f)) << 16);
  unsigned int w2 = (unsigned int)f2bf(fmaxf(b.x, 0.f)) | ((unsigned int)f2bf(fmaxf(b.y, 0.f)) << 16);
  unsigned int w3 = (unsigned int)f2bf(fmaxf(b.z, 0.f)) | ((unsigned int)f2bf(fmaxf(b.w, 0.f)) << 16);
  *(uint4*)(xh + e) = make_uint4(w0, w1, w2, w3);
}

// ---- prep: swizzle pointwise weights into WMMA B-fragment order -----------
// B frag (16x16x32 bf16): lane l holds column n = l&15, K = (l>>4)*16 + t,
// t = 0..15 contiguous per lane -> one 32B load per lane per fragment.
// wb[((nt*8 + kc)*32 + lane)*16 + t] = bf16(pw[(kc*32 + (lane>>4)*16 + t)*256 + nt*16 + (lane&15)])
__global__ __launch_bounds__(256) void build_wb(const float* __restrict__ pw,
                                                unsigned short* __restrict__ wb) {
  int f    = blockIdx.x * blockDim.x + threadIdx.x;   // 0..65535
  int t16  = f & 15;
  int lane = (f >> 4) & 31;
  int kc   = (f >> 9) & 7;
  int nt   = f >> 12;
  int k = kc * 32 + (lane >> 4) * 16 + t16;
  int n = nt * 16 + (lane & 15);
  wb[f] = f2bf(pw[k * CCH + n]);
}

// ---- prep: fold BN into scale/bias ----------------------------------------
__global__ void build_bn(const float* __restrict__ g, const float* __restrict__ b,
                         const float* __restrict__ m, const float* __restrict__ v,
                         float* __restrict__ s, float* __restrict__ o) {
  int c = threadIdx.x;
  float sc = g[c] * rsqrtf(v[c] + 1e-5f);
  s[c] = sc;
  o[c] = b[c] - m[c] * sc;
}

// ---- fused tile kernel -----------------------------------------------------
// PASS==1: gather xh(bf16) -> dwconv1 -> GEMM(pw1) -> bn1+relu -> h1 (bf16)
// PASS==2: gather h1(bf16) -> dwconv2 -> GEMM(pw2) -> bn2 + x  -> out (f32)
template <int PASS>
__global__ __launch_bounds__(256) void block_kernel(
    const unsigned short* __restrict__ src,   // bf16 gather source [N][256]
    const int*            __restrict__ nbr,   // [9][N]
    const float*          __restrict__ dw,    // [9][256] depthwise weights
    const unsigned short* __restrict__ wb,    // [65536] bf16 B-fragment layout
    const float*          __restrict__ bns,   // [256] folded BN scale
    const float*          __restrict__ bnb,   // [256] folded BN bias
    const float*          __restrict__ xskip, // pass2 skip input
    unsigned short*       __restrict__ out_bf,
    float*                __restrict__ out_f) {
  __shared__ __align__(32) unsigned short sA[8 * 32 * 16]; // A frags, 8 KB
  __shared__ float sD[TM * CCH];                           // D tile,  16 KB
  __shared__ float sW[KNBR * CCH];                         // dw wts,   9 KB
  __shared__ float sS[CCH], sB[CCH];

  const int t  = threadIdx.x;
  const int i0 = blockIdx.x * TM;

  for (int idx = t; idx < KNBR * CCH; idx += 256) sW[idx] = dw[idx];
  sS[t] = bns[t];
  sB[t] = bnb[t];
  __syncthreads();

  // ---- depthwise gather conv: thread owns (point i, channels c0..c0+15) ---
  const int i   = t >> 4;      // 0..15
  const int seg = t & 15;      // 0..15
  const int c0  = seg * 16;

  float acc[16];
#pragma unroll
  for (int j = 0; j < 16; ++j) acc[j] = 0.f;

#pragma unroll
  for (int k = 0; k < KNBR; ++k) {
    int r = nbr[k * NPTS + i0 + i];
    if (r < 0) continue;
    const uint4* p = (const uint4*)(src + (long long)r * CCH + c0);
    uint4 a = p[0], b = p[1];
    const float* w = &sW[k * CCH + c0];
    unsigned int av[4] = {a.x, a.y, a.z, a.w};
    unsigned int bv[4] = {b.x, b.y, b.z, b.w};
#pragma unroll
    for (int q = 0; q < 4; ++q) {
      acc[q * 2 + 0]     += bf2f((unsigned short)(av[q] & 0xffffu)) * w[q * 2 + 0];
      acc[q * 2 + 1]     += bf2f((unsigned short)(av[q] >> 16))     * w[q * 2 + 1];
      acc[8 + q * 2 + 0] += bf2f((unsigned short)(bv[q] & 0xffffu)) * w[8 + q * 2 + 0];
      acc[8 + q * 2 + 1] += bf2f((unsigned short)(bv[q] >> 16))     * w[8 + q * 2 + 1];
    }
  }

  // ---- pack into 16-bit A-fragment layout (contiguous k = 16h + slot, as wb) in LDS -------------
  // channel c = seg*16 + j;  kchunk = c>>5; lane = i + 16*((c>>4)&1);
  // slot = c&15  ->  two contiguous 16B stores per thread.
  {
    const int kchunk = seg >> 1;
    const int soff   = 16 * (seg & 1);
    unsigned int pk[4];
#pragma unroll
    for (int q = 0; q < 4; ++q)
      pk[q] = (unsigned int)f2bf(acc[q * 2]) | ((unsigned int)f2bf(acc[q * 2 + 1]) << 16);
    *(uint4*)&sA[(kchunk * 32 + soff + i) * 16] = make_uint4(pk[0], pk[1], pk[2], pk[3]);
#pragma unroll
    for (int q = 0; q < 4; ++q)
      pk[q] = (unsigned int)f2bf(acc[8 + q * 2]) | ((unsigned int)f2bf(acc[8 + q * 2 + 1]) << 16);
    *(uint4*)&sA[(kchunk * 32 + soff + i) * 16 + 8] = make_uint4(pk[0], pk[1], pk[2], pk[3]);
  }
  __syncthreads();

  // ---- GEMM: 8 waves x 2 N-tiles, 8 K-chunks of 32 -> 16 WMMA / wave ------
  {
    const int lane = t & 31;
    const int wv   = t >> 5;
    const int nt0  = wv * 2, nt1 = wv * 2 + 1;
    v8f cA = {}, cB = {};
#pragma unroll
    for (int kc = 0; kc < 8; ++kc) {
      v16bf af = *(const v16bf*)&sA[(kc * 32 + lane) * 16];
      v16bf b0 = *(const v16bf*)(wb + ((nt0 * 8 + kc) * 32 + lane) * 16);
      v16bf b1 = *(const v16bf*)(wb + ((nt1 * 8 + kc) * 32 + lane) * 16);
      cA = __builtin_amdgcn_wmma_f32_16x16x32_bf16(false, af, false, b0, (short)0, cA, false, false);
      cB = __builtin_amdgcn_wmma_f32_16x16x32_bf16(false, af, false, b1, (short)0, cB, false, false);
    }
    // D layout: VGPR r, lane l -> row m = r + 8*(l>>4), col = l&15
    const int m = 8 * (lane >> 4);
    const int n = lane & 15;
#pragma unroll
    for (int r = 0; r < 8; ++r) {
      sD[(m + r) * CCH + nt0 * 16 + n] = cA[r];
      sD[(m + r) * CCH + nt1 * 16 + n] = cB[r];
    }
  }
  __syncthreads();

  // ---- epilogue: BN (+relu->bf16 | +skip->f32), coalesced b128 stores -----
  const int e0 = t * 16;
  if (PASS == 1) {
    unsigned int pk[8];
#pragma unroll
    for (int j = 0; j < 16; j += 2) {
      int c = (e0 + j) & 255;
      float v0 = fmaxf(sD[e0 + j]     * sS[c]     + sB[c],     0.f);
      float v1 = fmaxf(sD[e0 + j + 1] * sS[c + 1] + sB[c + 1], 0.f);
      pk[j >> 1] = (unsigned int)f2bf(v0) | ((unsigned int)f2bf(v1) << 16);
    }
    uint4* dst = (uint4*)(out_bf + (long long)i0 * CCH + e0);
    dst[0] = make_uint4(pk[0], pk[1], pk[2], pk[3]);
    dst[1] = make_uint4(pk[4], pk[5], pk[6], pk[7]);
  } else {
    const float4* xs  = (const float4*)(xskip + (long long)i0 * CCH + e0);
    float4*       dst = (float4*)(out_f + (long long)i0 * CCH + e0);
#pragma unroll
    for (int q = 0; q < 4; ++q) {
      float4 xv = xs[q];
      int c = (e0 + q * 4) & 255;
      float4 o;
      o.x = sD[e0 + q * 4 + 0] * sS[c + 0] + sB[c + 0] + xv.x;
      o.y = sD[e0 + q * 4 + 1] * sS[c + 1] + sB[c + 1] + xv.y;
      o.z = sD[e0 + q * 4 + 2] * sS[c + 2] + sB[c + 2] + xv.z;
      o.w = sD[e0 + q * 4 + 3] * sS[c + 3] + sB[c + 3] + xv.w;
      dst[q] = o;
    }
  }
}

// ---------------------------------------------------------------------------
extern "C" void kernel_launch(void* const* d_in, const int* in_sizes, int n_in,
                              void* d_out, int out_size, void* d_ws, size_t ws_size,
                              hipStream_t stream) {
  (void)in_sizes; (void)n_in; (void)out_size; (void)ws_size;
  const float* x   = (const float*)d_in[0];
  const int*   nbr = (const int*)d_in[1];
  const float* dw1 = (const float*)d_in[2];
  const float* pw1 = (const float*)d_in[3];
  const float* g1  = (const float*)d_in[4];
  const float* b1  = (const float*)d_in[5];
  const float* m1  = (const float*)d_in[6];
  const float* v1  = (const float*)d_in[7];
  const float* dw2 = (const float*)d_in[8];
  const float* pw2 = (const float*)d_in[9];
  const float* g2  = (const float*)d_in[10];
  const float* b2  = (const float*)d_in[11];
  const float* m2  = (const float*)d_in[12];
  const float* v2  = (const float*)d_in[13];
  float* out = (float*)d_out;

  // workspace layout (bytes)
  char* ws = (char*)d_ws;
  unsigned short* xh  = (unsigned short*)(ws);                         // 128 MB
  unsigned short* h1  = (unsigned short*)(ws + 128000000ll);           // 128 MB
  unsigned short* wb1 = (unsigned short*)(ws + 256000000ll);           // 128 KB
  unsigned short* wb2 = (unsigned short*)(ws + 256000000ll + 131072);  // 128 KB
  float* s1 = (float*)(ws + 256000000ll + 2 * 131072);
  float* o1 = s1 + CCH;
  float* s2 = o1 + CCH;
  float* o2 = s2 + CCH;

  // prep
  relu_to_bf16<<<(NPTS * CCH) / (256 * 8), 256, 0, stream>>>(x, xh);   // 31250 blocks
  build_wb<<<256, 256, 0, stream>>>(pw1, wb1);
  build_wb<<<256, 256, 0, stream>>>(pw2, wb2);
  build_bn<<<1, 256, 0, stream>>>(g1, b1, m1, v1, s1, o1);
  build_bn<<<1, 256, 0, stream>>>(g2, b2, m2, v2, s2, o2);

  // fused passes
  block_kernel<1><<<NTILES, 256, 0, stream>>>(xh, nbr, dw1, wb1, s1, o1, nullptr, h1, nullptr);
  block_kernel<2><<<NTILES, 256, 0, stream>>>(h1, nbr, dw2, wb2, s2, o2, x, nullptr, out);
}
